// MockHilbertRingAttention_51531017617710
// MI455X (gfx1250) — matrix operand order repaired and checked by reading, not compiled, not run on hardware
//
#include <hip/hip_runtime.h>
#include <hip/hip_bf16.h>

typedef __attribute__((ext_vector_type(16))) __bf16 v16bf;
typedef __attribute__((ext_vector_type(8)))  __bf16 v8bf;
typedef __attribute__((ext_vector_type(8)))  float  v8f;
typedef unsigned short ushort_t;
typedef int v4i_t __attribute__((vector_size(16)));   // matches builtin param type

#define HIDDEN 2048
#define SEQ    8192
#define CHUNK  2048
#define RING   4
#define BATCH  2
#define GRID91 91

#define AS1 __attribute__((address_space(1)))
#define AS3 __attribute__((address_space(3)))

#if __has_builtin(__builtin_amdgcn_global_load_async_to_lds_b128)
#define USE_ASYNC_LDS 1
#else
#define USE_ASYNC_LDS 0
#endif

__device__ __forceinline__ void wait_async0() {
#if __has_builtin(__builtin_amdgcn_s_wait_asynccnt)
    __builtin_amdgcn_s_wait_asynccnt(0);
#else
    asm volatile("s_wait_asynccnt 0x0" ::: "memory");
#endif
}

// Snake (boustrophedon) mapping over a 91x91 grid; involution, and it maps
// [0,2048) onto itself (rows 0..21 full, row 22 even => identity).
__device__ __forceinline__ int snake_map(int p) {
    int row = p / GRID91;
    int col = p - row * GRID91;
    return (row & 1) ? (row * GRID91 + (GRID91 - 1 - col)) : p;
}

__device__ __forceinline__ ushort_t f2bf(float f) {
    unsigned u = __float_as_uint(f);
    u += 0x7FFFu + ((u >> 16) & 1u);   // round-to-nearest-even
    return (ushort_t)(u >> 16);
}

// ---------------- 1) zero only rows [CHUNK, SEQ) of the output ----------------
__global__ void zero_tail_kernel(float4* __restrict__ out4) {
    const size_t perBatch = (size_t)(SEQ - CHUNK) * HIDDEN / 4;  // tail float4s per batch
    size_t i = (size_t)blockIdx.x * blockDim.x + threadIdx.x;
    size_t b = i / perBatch;
    size_t loc = i - b * perBatch;
    out4[b * ((size_t)SEQ * HIDDEN / 4) + ((size_t)CHUNK * HIDDEN / 4) + loc] =
        make_float4(0.f, 0.f, 0.f, 0.f);
}

// ---------------- 2) softmax over rows + ring-sum -> bf16 A ----------------
// grid (CHUNK, BATCH), block 256.  A[b,i,j] = sum_r softmax(scores[r,b,i,:]*scale)[j]
__global__ void softmax_sum_kernel(const float* __restrict__ scores,
                                   ushort_t* __restrict__ A) {
    __shared__ float red[256];
    const int i = blockIdx.x;
    const int b = blockIdx.y;
    const int t = threadIdx.x;
    const float scale = 0.088388347648318447f;  // 1/sqrt(128)

    float acc[8] = {0.f, 0.f, 0.f, 0.f, 0.f, 0.f, 0.f, 0.f};

    for (int r = 0; r < RING; ++r) {
        const float* row = scores + (((size_t)(r * BATCH + b) * CHUNK + i) * CHUNK);
        float4 v0 = *(const float4*)(row + 4 * t);
        float4 v1 = *(const float4*)(row + 1024 + 4 * t);
        float e[8] = {v0.x, v0.y, v0.z, v0.w, v1.x, v1.y, v1.z, v1.w};
        float m = -3.4e38f;
        #pragma unroll
        for (int k = 0; k < 8; ++k) { e[k] *= scale; m = fmaxf(m, e[k]); }
        red[t] = m; __syncthreads();
        for (int off = 128; off > 0; off >>= 1) {
            if (t < off) red[t] = fmaxf(red[t], red[t + off]);
            __syncthreads();
        }
        m = red[0]; __syncthreads();
        float s = 0.f;
        #pragma unroll
        for (int k = 0; k < 8; ++k) { e[k] = __expf(e[k] - m); s += e[k]; }
        red[t] = s; __syncthreads();
        for (int off = 128; off > 0; off >>= 1) {
            if (t < off) red[t] += red[t + off];
            __syncthreads();
        }
        s = red[0]; __syncthreads();
        float inv = 1.0f / s;
        #pragma unroll
        for (int k = 0; k < 8; ++k) acc[k] += e[k] * inv;
    }

    ushort_t* dst = A + ((size_t)b * CHUNK + i) * CHUNK;
    *(ushort4*)(dst + 4 * t) =
        make_ushort4(f2bf(acc[0]), f2bf(acc[1]), f2bf(acc[2]), f2bf(acc[3]));
    *(ushort4*)(dst + 1024 + 4 * t) =
        make_ushort4(f2bf(acc[4]), f2bf(acc[5]), f2bf(acc[6]), f2bf(acc[7]));
}

// ------- 3) gather + transpose pack: Bt[b,d,j] = bf16(x[b,m(j),d]) -------
// grid (64 j-tiles, 64 d-tiles, BATCH), block (32,8)
__global__ void pack_bt_kernel(const float* __restrict__ x,
                               ushort_t* __restrict__ Bt) {
    __shared__ float tile[32][33];
    const int jt = blockIdx.x, dt = blockIdx.y, b = blockIdx.z;
    const int tx = threadIdx.x, ty = threadIdx.y;
    #pragma unroll
    for (int i = 0; i < 4; ++i) {
        int jr = ty + i * 8;
        int j  = jt * 32 + jr;
        int gr = snake_map(j);                       // gathered x row
        tile[jr][tx] = x[((size_t)b * SEQ + gr) * HIDDEN + dt * 32 + tx];
    }
    __syncthreads();
    #pragma unroll
    for (int i = 0; i < 4; ++i) {
        int dr = ty + i * 8;
        int d  = dt * 32 + dr;
        int j  = jt * 32 + tx;
        Bt[((size_t)b * HIDDEN + d) * CHUNK + j] = f2bf(tile[tx][dr]);
    }
}

// ---------------- 4) bf16 WMMA GEMM (double-buffered LDS) + snake scatter ----------------
// acc[b] = A[b] (2048x2048) x B[b] (2048x2048); out[b, m(i), d] = acc[b,i,d]
// grid (16 n-tiles, 16 m-tiles, BATCH), block 256 = 8 waves.
__global__ void __launch_bounds__(256)
gemm_scatter_kernel(const ushort_t* __restrict__ A,
                    const ushort_t* __restrict__ Bt,
                    float* __restrict__ out) {
    __shared__ ushort_t sA[2][128][40];   // double-buffered 128(M) x 32(K), padded
    __shared__ ushort_t sB[2][128][40];   // double-buffered 128(N) x 32(K), padded

    const int nt0 = blockIdx.x * 128;
    const int mt0 = blockIdx.y * 128;
    const int b   = blockIdx.z;
    const int t   = threadIdx.x;
    const int lane   = t & 31;
    const int wid    = t >> 5;
    const int laneLo = lane & 15;
    const int khalf  = lane >> 4;
    const int wrow = (wid >> 2) * 64;   // wave M offset
    const int wcol = (wid & 3) * 32;    // wave N offset

    const ushort_t* gA = A  + (size_t)b * CHUNK * CHUNK;
    const ushort_t* gB = Bt + (size_t)b * HIDDEN * CHUNK;

    // staging assignment: 2 x b128 per thread per matrix
    const int row0 = t >> 2,          seg0 = t & 3;
    const int row1 = (t + 256) >> 2,  seg1 = (t + 256) & 3;

    v8f acc[4][2];
    #pragma unroll
    for (int mi = 0; mi < 4; ++mi)
        #pragma unroll
        for (int ni = 0; ni < 2; ++ni)
            acc[mi][ni] = (v8f){0.f, 0.f, 0.f, 0.f, 0.f, 0.f, 0.f, 0.f};

#if USE_ASYNC_LDS
    #define STAGE_TILE(buf, kk)                                                          \
        do {                                                                             \
            __builtin_amdgcn_global_load_async_to_lds_b128(                              \
                (AS1 v4i_t*)(gA + (size_t)(mt0 + row0) * CHUNK + (kk) + seg0 * 8),       \
                (AS3 v4i_t*)&sA[(buf)][row0][seg0 * 8], 0, 0);                           \
            __builtin_amdgcn_global_load_async_to_lds_b128(                              \
                (AS1 v4i_t*)(gA + (size_t)(mt0 + row1) * CHUNK + (kk) + seg1 * 8),       \
                (AS3 v4i_t*)&sA[(buf)][row1][seg1 * 8], 0, 0);                           \
            __builtin_amdgcn_global_load_async_to_lds_b128(                              \
                (AS1 v4i_t*)(gB + (size_t)(nt0 + row0) * CHUNK + (kk) + seg0 * 8),       \
                (AS3 v4i_t*)&sB[(buf)][row0][seg0 * 8], 0, 0);                           \
            __builtin_amdgcn_global_load_async_to_lds_b128(                              \
                (AS1 v4i_t*)(gB + (size_t)(nt0 + row1) * CHUNK + (kk) + seg1 * 8),       \
                (AS3 v4i_t*)&sB[(buf)][row1][seg1 * 8], 0, 0);                           \
        } while (0)

    STAGE_TILE(0, 0);
    wait_async0();
    __syncthreads();
#else
    {   // synchronous preload of tile 0
        int4 a0 = *(const int4*)(gA + (size_t)(mt0 + row0) * CHUNK + seg0 * 8);
        int4 a1 = *(const int4*)(gA + (size_t)(mt0 + row1) * CHUNK + seg1 * 8);
        int4 b0 = *(const int4*)(gB + (size_t)(nt0 + row0) * CHUNK + seg0 * 8);
        int4 b1 = *(const int4*)(gB + (size_t)(nt0 + row1) * CHUNK + seg1 * 8);
        *(int4*)(&sA[0][row0][seg0 * 8]) = a0;
        *(int4*)(&sA[0][row1][seg1 * 8]) = a1;
        *(int4*)(&sB[0][row0][seg0 * 8]) = b0;
        *(int4*)(&sB[0][row1][seg1 * 8]) = b1;
        __syncthreads();
    }
#endif

    for (int k0 = 0; k0 < CHUNK; k0 += 32) {
        const int p = (k0 >> 5) & 1;
        const bool has_next = (k0 + 32) < CHUNK;

#if USE_ASYNC_LDS
        if (has_next) STAGE_TILE(p ^ 1, k0 + 32);   // overlaps with WMMA below
#else
        int4 a0, a1, b0, b1;
        if (has_next) {   // issue next-tile global loads early; overlap with WMMA
            a0 = *(const int4*)(gA + (size_t)(mt0 + row0) * CHUNK + k0 + 32 + seg0 * 8);
            a1 = *(const int4*)(gA + (size_t)(mt0 + row1) * CHUNK + k0 + 32 + seg1 * 8);
            b0 = *(const int4*)(gB + (size_t)(nt0 + row0) * CHUNK + k0 + 32 + seg0 * 8);
            b1 = *(const int4*)(gB + (size_t)(nt0 + row1) * CHUNK + k0 + 32 + seg1 * 8);
        }
#endif
        // WGP-scope prefetch two tiles ahead
        if (k0 + 64 < CHUNK && t < 128) {
            __builtin_prefetch(gA + (size_t)(mt0 + t) * CHUNK + k0 + 64, 0, 3);
            __builtin_prefetch(gB + (size_t)(nt0 + t) * CHUNK + k0 + 64, 0, 3);
        }

        // A fragment (16x32 bf16): lanes<16 hold K0-7 (lo) & K16-23 (hi); lanes>=16 K8-15 & K24-31
        v16bf af[4];
        #pragma unroll
        for (int mi = 0; mi < 4; ++mi) {
            const ushort_t* rp = &sA[p][wrow + mi * 16 + laneLo][0];
            v8bf lo = *(const v8bf*)(rp + khalf * 8);
            v8bf hi = *(const v8bf*)(rp + 16 + khalf * 8);
            af[mi] = __builtin_shufflevector(lo, hi, 0, 1, 2, 3, 4, 5, 6, 7,
                                             8, 9, 10, 11, 12, 13, 14, 15);
        }
        // B fragment (32x16 bf16): same K split as A (lanes<16 K0-7 & K16-23; lanes>=16 K8-15 & K24-31)
        v16bf bfr[2];
        #pragma unroll
        for (int ni = 0; ni < 2; ++ni) {
            const ushort_t* cp = &sB[p][wcol + ni * 16 + laneLo][0];
            v8bf lo = *(const v8bf*)(cp + khalf * 8);
            v8bf hi = *(const v8bf*)(cp + 16 + khalf * 8);
            bfr[ni] = __builtin_shufflevector(lo, hi, 0, 1, 2, 3, 4, 5, 6, 7,
                                              8, 9, 10, 11, 12, 13, 14, 15);
        }

        #pragma unroll
        for (int mi = 0; mi < 4; ++mi)
            #pragma unroll
            for (int ni = 0; ni < 2; ++ni)
                acc[mi][ni] = __builtin_amdgcn_wmma_f32_16x16x32_bf16(
                    false, af[mi], false, bfr[ni], (short)0, acc[mi][ni],
                    false, false);

#if USE_ASYNC_LDS
        if (has_next) wait_async0();
#else
        if (has_next) {
            *(int4*)(&sA[p ^ 1][row0][seg0 * 8]) = a0;
            *(int4*)(&sA[p ^ 1][row1][seg1 * 8]) = a1;
            *(int4*)(&sB[p ^ 1][row0][seg0 * 8]) = b0;
            *(int4*)(&sB[p ^ 1][row1][seg1 * 8]) = b1;
        }
#endif
        __syncthreads();
    }

    // C layout: VGPR v, lanes 0-15 => M=v, N=lane; lanes 16-31 => M=v+8, N=lane-16
    float* ob = out + (size_t)b * SEQ * HIDDEN;
    #pragma unroll
    for (int mi = 0; mi < 4; ++mi) {
        #pragma unroll
        for (int ni = 0; ni < 2; ++ni) {
            #pragma unroll
            for (int v = 0; v < 8; ++v) {
                int gi = mt0 + wrow + mi * 16 + khalf * 8 + v;
                int gd = nt0 + wcol + ni * 16 + laneLo;
                int orow = snake_map(gi);   // stays in [0, 2048)
                ob[(size_t)orow * HIDDEN + gd] = acc[mi][ni][v];
            }
        }
    }
}

extern "C" void kernel_launch(void* const* d_in, const int* in_sizes, int n_in,
                              void* d_out, int out_size, void* d_ws, size_t ws_size,
                              hipStream_t stream) {
    const float* x      = (const float*)d_in[0];
    const float* scores = (const float*)d_in[1];
    float*       out    = (float*)d_out;

    ushort_t* A  = (ushort_t*)d_ws;                        // 2*2048*2048 bf16 = 16 MB
    ushort_t* Bt = A + (size_t)BATCH * CHUNK * CHUNK;      // 2*2048*2048 bf16 = 16 MB

    (void)in_sizes; (void)n_in; (void)out_size; (void)ws_size;

    // 1) zero only rows [2048, 8192): 2 * 6144 * 2048 floats = 6,291,456 float4
    zero_tail_kernel<<<dim3((unsigned)((size_t)BATCH * (SEQ - CHUNK) * HIDDEN / 4 / 256)),
                       256, 0, stream>>>((float4*)out);
    // 2) softmax + ring sum -> A (bf16)
    softmax_sum_kernel<<<dim3(CHUNK, BATCH), 256, 0, stream>>>(scores, A);
    // 3) gather + transpose pack -> Bt (bf16)
    pack_bt_kernel<<<dim3(CHUNK / 32, HIDDEN / 32, BATCH), dim3(32, 8), 0, stream>>>(x, Bt);
    // 4) WMMA GEMM + snake scatter (covers rows [0, 2048) completely)
    gemm_scatter_kernel<<<dim3(HIDDEN / 128, CHUNK / 128, BATCH), 256, 0, stream>>>(A, Bt, out);
}
